// MultiHeadPreformerAttention_50946902065475
// MI455X (gfx1250) — compile-verified, matrix operand order checked by reading, not run on hardware
//
#include <hip/hip_runtime.h>
#include <math.h>

typedef __attribute__((ext_vector_type(16))) _Float16 v16h;
typedef __attribute__((ext_vector_type(8)))  float    v8f;
typedef __attribute__((ext_vector_type(8)))  _Float16 h8v;
typedef __attribute__((ext_vector_type(4)))  _Float16 h4v;

#define BM 128
#define BN 128
#define BK 32
#define LDSS 40  // BK + 8 halfs of padding

// ---------------------------------------------------------------------------
// Async global->LDS copy (CDNA5, ASYNCcnt-tracked). 16 bytes per lane.
// ---------------------------------------------------------------------------
__device__ __forceinline__ void async_ld_b128(unsigned lds_off, const void* gptr) {
  asm volatile("global_load_async_to_lds_b128 %0, %1, off"
               :: "v"(lds_off), "v"(gptr) : "memory");
}
__device__ __forceinline__ void wait_async0() {
  asm volatile("s_wait_asynccnt 0x0" ::: "memory");
}

// ---------------------------------------------------------------------------
// Fragment gather from LDS per CDNA5 wave32 16-bit A/B layout (ISA 7.12.2):
// lanes 0-15: halfs[0..7]=K0..7, halfs[8..15]=K16..23 ; lanes 16-31: +8.
// ---------------------------------------------------------------------------
__device__ __forceinline__ v16h frag_from_lds(const _Float16* base, int row0, int lane) {
  const int r  = row0 + (lane & 15);
  const int kb = (lane & 16) ? 8 : 0;
  const _Float16* p = base + r * LDSS + kb;
  h8v lo = *(const h8v*)(p);
  h8v hi = *(const h8v*)(p + 16);
  v16h f;
#pragma unroll
  for (int i = 0; i < 8; ++i) { f[i] = lo[i]; f[8 + i] = hi[i]; }
  return f;
}

// ---------------------------------------------------------------------------
// C[M,N](f32) = scale*A(f32) @ Bt(f16,[N][K])ᵀ + bias, via WMMA f16.
// A tile staged through VALU (f32->f16 convert); B tile staged with async
// global->LDS b128 copies. blockIdx.z selects an (a_zoff, c_zoff) slice.
// bias must be non-null (pass a zero buffer for "no bias").
// ---------------------------------------------------------------------------
__global__ __launch_bounds__(256)
void gemm_wmma_f16(const float* __restrict__ A, int lda, int a_zoff,
                   const _Float16* __restrict__ Bt,
                   const float* __restrict__ bias,
                   float* __restrict__ C, int ldc, int c_zoff,
                   int K, float a_scale)
{
  __shared__ alignas(16) _Float16 As[BM * LDSS];
  __shared__ alignas(16) _Float16 Bs[BN * LDSS];

  const int n0   = blockIdx.x * BN;
  const int m0   = blockIdx.y * BM;
  const int tid  = threadIdx.x;
  const int lane = tid & 31;
  const int wave = tid >> 5;
  const int wm   = wave >> 2;  // 0..1  (64 rows each)
  const int wn   = wave & 3;   // 0..3  (32 cols each)

  const float* Az = A + (size_t)blockIdx.z * a_zoff;
  float*       Cz = C + (size_t)blockIdx.z * c_zoff;

  v8f acc[4][2];
#pragma unroll
  for (int i = 0; i < 4; ++i)
#pragma unroll
    for (int j = 0; j < 2; ++j) acc[i][j] = v8f{};

  for (int k0 = 0; k0 < K; k0 += BK) {
    // B tile: 128x32 halfs = 512 x b128, async global->LDS (no VGPR staging)
#pragma unroll
    for (int it = 0; it < 2; ++it) {
      const int idx = tid + it * 256;   // 0..511
      const int n   = idx >> 2;         // 0..127
      const int c   = (idx & 3) << 3;   // half index: 0,8,16,24
      async_ld_b128((unsigned)(size_t)&Bs[n * LDSS + c],
                    Bt + (size_t)(n0 + n) * K + k0 + c);
    }
    // A tile: 128x32 f32 -> f16*scale through VGPRs (needs the convert)
#pragma unroll
    for (int it = 0; it < 4; ++it) {
      const int idx = tid + it * 256;
      const int row = idx >> 3;
      const int c4  = (idx & 7) << 2;
      const float4 x = *(const float4*)(Az + (size_t)(m0 + row) * lda + k0 + c4);
      h4v hh;
      hh[0] = (_Float16)(x.x * a_scale);
      hh[1] = (_Float16)(x.y * a_scale);
      hh[2] = (_Float16)(x.z * a_scale);
      hh[3] = (_Float16)(x.w * a_scale);
      *(h4v*)(&As[row * LDSS + c4]) = hh;
    }
    wait_async0();
    __syncthreads();

    v16h afr[4], bfr[2];
#pragma unroll
    for (int i = 0; i < 4; ++i) afr[i] = frag_from_lds(As, wm * 64 + 16 * i, lane);
#pragma unroll
    for (int j = 0; j < 2; ++j) bfr[j] = frag_from_lds(Bs, wn * 32 + 16 * j, lane);

#pragma unroll
    for (int i = 0; i < 4; ++i)
#pragma unroll
      for (int j = 0; j < 2; ++j)
        acc[i][j] = __builtin_amdgcn_wmma_f32_16x16x32_f16(
            false, afr[i], false, bfr[j], (short)0, acc[i][j], false, false);

    __syncthreads();
  }

  // Epilogue: C/D layout (ISA 7.12.2): lane&15 -> N, lane>=16 -> M+8, VGPR r -> M+r
#pragma unroll
  for (int i = 0; i < 4; ++i) {
    const int mg = m0 + wm * 64 + 16 * i + ((lane & 16) ? 8 : 0);
#pragma unroll
    for (int j = 0; j < 2; ++j) {
      const int ng = n0 + wn * 32 + 16 * j + (lane & 15);
      const float bz = bias[ng];
#pragma unroll
      for (int r = 0; r < 8; ++r)
        Cz[(size_t)(mg + r) * ldc + ng] = acc[i][j][r] + bz;
    }
  }
}

// Wt[n*K + k] = (f16) W[k*N + n]   (W row-major [K][N])
__global__ void wtrans_f16(const float* __restrict__ W, _Float16* __restrict__ Wt,
                           int K, int N) {
  const int idx = blockIdx.x * blockDim.x + threadIdx.x;
  if (idx >= K * N) return;
  const int n = idx / K, k = idx - n * K;
  Wt[idx] = (_Float16)W[(size_t)k * N + n];
}

__global__ void cvt_f16(const float* __restrict__ src, _Float16* __restrict__ dst, int n) {
  const int i = blockIdx.x * blockDim.x + threadIdx.x;
  if (i < n) dst[i] = (_Float16)src[i];
}

__global__ void fillzero(float* __restrict__ p, int n) {
  const int i = blockIdx.x * blockDim.x + threadIdx.x;
  if (i < n) p[i] = 0.0f;
}

// diag[row] = scl * sum_d proj_in[row*512 + d]^2 ; one wave per row, 8 rows/block
__global__ __launch_bounds__(256)
void diag_kernel(const float* __restrict__ pin, float* __restrict__ diag, float scl) {
  const int row  = blockIdx.x * 8 + (threadIdx.x >> 5);
  const int lane = threadIdx.x & 31;
  const float* p = pin + (size_t)row * 512;
  float s = 0.f;
#pragma unroll
  for (int i = 0; i < 16; ++i) { const float x = p[lane + 32 * i]; s += x * x; }
#pragma unroll
  for (int off = 16; off > 0; off >>= 1) s += __shfl_xor(s, off, 32);
  if (lane == 0) diag[row] = s * scl;
}

// q' = ratio*(exp(dd - diag - rowmax(dd)) + eps), in place; one wave per 256-row
__global__ __launch_bounds__(256)
void qprime_kernel(float* __restrict__ dd, const float* __restrict__ diag,
                   float ratio, float eps) {
  const int row  = blockIdx.x * 8 + (threadIdx.x >> 5);
  const int lane = threadIdx.x & 31;
  float* p = dd + (size_t)row * 256;
  float v[8]; float mx = -3.4e38f;
#pragma unroll
  for (int i = 0; i < 8; ++i) { v[i] = p[lane + 32 * i]; mx = fmaxf(mx, v[i]); }
#pragma unroll
  for (int off = 16; off > 0; off >>= 1) mx = fmaxf(mx, __shfl_xor(mx, off, 32));
  const float dg = diag[row];
#pragma unroll
  for (int i = 0; i < 8; ++i) p[lane + 32 * i] = ratio * (__expf(v[i] - dg - mx) + eps);
}

// global max over (L, m) per (b,h): one 1024-thread block per (b,h)
__global__ __launch_bounds__(1024)
void kmax_kernel(const float* __restrict__ ddk, float* __restrict__ kmax) {
  const int b = blockIdx.x >> 1, h = blockIdx.x & 1;
  const float* base = ddk + (size_t)b * 4096 * 512 + h * 256;
  float mx = -3.4e38f;
  for (int idx = threadIdx.x; idx < 4096 * 256; idx += 1024) {
    const int l = idx >> 8, m = idx & 255;
    mx = fmaxf(mx, base[(size_t)l * 512 + m]);
  }
#pragma unroll
  for (int off = 16; off > 0; off >>= 1) mx = fmaxf(mx, __shfl_xor(mx, off, 32));
  __shared__ float red[32];
  if ((threadIdx.x & 31) == 0) red[threadIdx.x >> 5] = mx;
  __syncthreads();
  if (threadIdx.x < 32) {
    mx = red[threadIdx.x];
#pragma unroll
    for (int off = 16; off > 0; off >>= 1) mx = fmaxf(mx, __shfl_xor(mx, off, 32));
    if (threadIdx.x == 0) kmax[blockIdx.x] = mx;
  }
}

__global__ __launch_bounds__(256)
void kprime_kernel(float* __restrict__ dd, const float* __restrict__ diag,
                   const float* __restrict__ kmax, float ratio, float eps) {
  const int row  = blockIdx.x * 8 + (threadIdx.x >> 5);
  const int lane = threadIdx.x & 31;
  float* p = dd + (size_t)row * 256;
  const float mx = kmax[((row >> 13) << 1) | (row & 1)];
  const float dg = diag[row];
#pragma unroll
  for (int i = 0; i < 8; ++i) {
    const float v = p[lane + 32 * i];
    p[lane + 32 * i] = ratio * (__expf(v - dg - mx) + eps);
  }
}

// Per (seq,h): S = Q'K'T (4x4 over m=256), norm = rowsum(S), out = S@V / norm,
// written directly into the reference's transpose/reshape ("concat") layout.
__global__ __launch_bounds__(128)
void attn_core(const float* __restrict__ qp, const float* __restrict__ kp,
               const float* __restrict__ vproj, float* __restrict__ concat) {
  const int seq = blockIdx.x, h = blockIdx.y;
  const int wave = threadIdx.x >> 5, lane = threadIdx.x & 31;
  __shared__ float S[4][4];
  __shared__ float nrm[4];
  {
    const float* qrow = qp + ((size_t)(wave * 4096 + seq) * 2 + h) * 256;
    float sij[4] = {0.f, 0.f, 0.f, 0.f};
    for (int m = lane; m < 256; m += 32) {
      const float qv = qrow[m];
#pragma unroll
      for (int j = 0; j < 4; ++j)
        sij[j] += qv * kp[((size_t)(j * 4096 + seq) * 2 + h) * 256 + m];
    }
#pragma unroll
    for (int j = 0; j < 4; ++j)
#pragma unroll
      for (int off = 16; off > 0; off >>= 1) sij[j] += __shfl_xor(sij[j], off, 32);
    if (lane == 0) {
      float t = 0.f;
#pragma unroll
      for (int j = 0; j < 4; ++j) { S[wave][j] = sij[j]; t += sij[j]; }
      nrm[wave] = t;
    }
  }
  __syncthreads();
  for (int t = threadIdx.x; t < 2048; t += 128) {
    const int i = t >> 9, d = t & 511;
    float acc = 0.f;
#pragma unroll
    for (int j = 0; j < 4; ++j)
      acc += S[i][j] * vproj[(size_t)(j * 4096 + seq) * 1024 + h * 512 + d];
    // concat[b_out, l_out, c_out]: b_out=seq>>10, l_out=(seq&1023)*4+2h+(i>>1),
    // c_out=(i&1)*512+d  (closed form of transpose(0,2,1,3)+reshape)
    const size_t co = ((size_t)(seq >> 10)) * (4096u * 1024u)
                    + ((size_t)((seq & 1023) * 4 + h * 2 + (i >> 1))) * 1024u
                    + (size_t)((i & 1) * 512 + d);
    concat[co] = acc / nrm[i];
  }
}

// ---------------------------------------------------------------------------
extern "C" void kernel_launch(void* const* d_in, const int* in_sizes, int n_in,
                              void* d_out, int out_size, void* d_ws, size_t ws_size,
                              hipStream_t stream) {
  (void)in_sizes; (void)n_in; (void)out_size; (void)ws_size;
  const float* q_in  = (const float*)d_in[0];
  const float* k_in  = (const float*)d_in[1];
  const float* v_in  = (const float*)d_in[2];
  // d_in[3] = mask (unused by reference)
  const float* Wq = (const float*)d_in[4];  const float* bq = (const float*)d_in[5];
  const float* Wk = (const float*)d_in[6];  const float* bk = (const float*)d_in[7];
  const float* Wv = (const float*)d_in[8];  const float* bv = (const float*)d_in[9];
  const float* Wo = (const float*)d_in[10]; const float* bo = (const float*)d_in[11];
  const float* feats = (const float*)d_in[12];
  float* out = (float*)d_out;

  char* ws = (char*)d_ws;
  const size_t MB = 1024u * 1024u;
  float* qproj = (float*)(ws + 0);            // 64 MB
  float* kproj = (float*)(ws + 64 * MB);      // 64 MB
  float* vproj = (float*)(ws + 128 * MB);     // 64 MB
  float* ddq   = (float*)(ws + 192 * MB);     // 32 MB  [bl][h][m] = q'
  float* ddk   = (float*)(ws + 224 * MB);     // 32 MB  = k'
  float* diagq = (float*)(ws + 256 * MB);             // 128 KB
  float* diagk = (float*)(ws + 256 * MB + 131072);    // 128 KB
  float* kmax  = (float*)(ws + 256 * MB + 262144);    // 8 floats
  float* zbias = (float*)(ws + 256 * MB + 262400);    // 1024 floats of zero
  _Float16* Wtq   = (_Float16*)(ws + 256 * MB + 266496);
  _Float16* Wtk   = Wtq + 1024 * 1024;
  _Float16* Wtv   = Wtk + 1024 * 1024;
  _Float16* Wto   = Wtv + 1024 * 1024;
  _Float16* projh = Wto + 1024 * 1024;        // 256x512 f16
  float* concat = qproj;                      // reuse: qproj dead after ddq/diagq

  const float s = 1.0f / sqrtf(sqrtf(512.0f));       // d^-1/4
  const float diag_scale = 0.5f * s * s;             // 0.5 / sqrt(512)
  const float ratio = 1.0f / 16.0f;                  // 1/sqrt(M)
  const float eps = 1e-6f;

  // Weight prep (f32 -> f16, transposed to [N][K]) + zero bias
  wtrans_f16<<<4096, 256, 0, stream>>>(Wq, Wtq, 1024, 1024);
  wtrans_f16<<<4096, 256, 0, stream>>>(Wk, Wtk, 1024, 1024);
  wtrans_f16<<<4096, 256, 0, stream>>>(Wv, Wtv, 1024, 1024);
  wtrans_f16<<<4096, 256, 0, stream>>>(Wo, Wto, 1024, 1024);
  cvt_f16<<<512, 256, 0, stream>>>(feats, projh, 256 * 512);
  fillzero<<<4, 256, 0, stream>>>(zbias, 1024);

  // Input projections: [16384,1024] @ [1024,1024] + b
  dim3 g1(1024 / BN, 16384 / BM, 1);
  gemm_wmma_f16<<<g1, 256, 0, stream>>>(q_in, 1024, 0, Wtq, bq, qproj, 1024, 0, 1024, 1.0f);
  gemm_wmma_f16<<<g1, 256, 0, stream>>>(k_in, 1024, 0, Wtk, bk, kproj, 1024, 0, 1024, 1.0f);
  gemm_wmma_f16<<<g1, 256, 0, stream>>>(v_in, 1024, 0, Wtv, bv, vproj, 1024, 0, 1024, 1.0f);

  // Feature maps: dd[bl,h,m] = (s * proj_rows) @ featsT, per head via blockIdx.z
  dim3 g2(256 / BN, 16384 / BM, 2);
  gemm_wmma_f16<<<g2, 256, 0, stream>>>(qproj, 1024, 512, projh, zbias, ddq, 512, 256, 512, s);
  gemm_wmma_f16<<<g2, 256, 0, stream>>>(kproj, 1024, 512, projh, zbias, ddk, 512, 256, 512, s);

  diag_kernel<<<4096, 256, 0, stream>>>(qproj, diagq, diag_scale);
  diag_kernel<<<4096, 256, 0, stream>>>(kproj, diagk, diag_scale);

  qprime_kernel<<<4096, 256, 0, stream>>>(ddq, diagq, ratio, eps);
  kmax_kernel<<<8, 1024, 0, stream>>>(ddk, kmax);
  kprime_kernel<<<4096, 256, 0, stream>>>(ddk, diagk, kmax, ratio, eps);

  // Tiny (Q'K'T)V attention core, writes the "concat" layout directly
  dim3 g3(4096, 2, 1);
  attn_core<<<g3, 128, 0, stream>>>(ddq, ddk, vproj, concat);

  // Output projection
  gemm_wmma_f16<<<g1, 256, 0, stream>>>(concat, 1024, 0, Wto, bo, out, 1024, 0, 1024, 1.0f);
}
